// GraphAttentionLayer_18073222381748
// MI455X (gfx1250) — matrix operand order repaired and checked by reading, not compiled, not run on hardware
//
#include <hip/hip_runtime.h>

// ---------------------------------------------------------------------------
// Linear-attention graph layer, fused bf16-WMMA pipeline for gfx1250 (wave32).
//   Q = elu(h@Wq)+1 ; K = elu(h@Wk)+1 ; V = h@Wv
//   KV = K^T V ; ksum = sum_rows(K) ; Z = 1/(Q ksum) ; out = elu((Q@KV)*Z)
// N=200000, D_IN=512, D_OUT=128.  All GEMMs: v_wmma_f32_16x16x32_bf16.
// ---------------------------------------------------------------------------

typedef __bf16        v16bf __attribute__((ext_vector_type(16)));
typedef float         v8f   __attribute__((ext_vector_type(8)));
typedef unsigned int  ui4   __attribute__((ext_vector_type(4)));
typedef float         f4    __attribute__((ext_vector_type(4)));

union Frag { ui4 u[2]; v16bf v; };   // 32B: one WMMA A/B fragment per lane

static constexpr int    N_ROWS  = 200000;
static constexpr int    D_IN    = 512;
static constexpr int    D_OUT   = 128;
static constexpr int    CHUNK   = 64;               // rows per block iteration
static constexpr int    NCHUNK  = N_ROWS / CHUNK;   // 3125 (exact)
static constexpr int    NBLK1   = 512;              // persistent blocks, phase 1
static constexpr int    LDSTR   = 72;               // halfword stride: 144B = 9*16 (aligned) & bank-staggered

// workspace layout (bytes)
static constexpr size_t WS_WT   = 0;                                              // bf16 Wt[3][128][512]
static constexpr size_t WS_KVT  = WS_WT  + (size_t)3*128*512*2;                   // bf16 KVt[128][128] (n-major)
static constexpr size_t WS_KSUM = WS_KVT + (size_t)128*128*2;                     // f32  ksum[128]
static constexpr size_t WS_Z    = WS_KSUM + 512;                                  // f32  Z[N]
static constexpr size_t WS_Q    = (WS_Z + (size_t)N_ROWS*4 + 255) & ~(size_t)255; // bf16 Q[N][128] row-major
static constexpr size_t WS_KVP  = (WS_Q + (size_t)N_ROWS*D_OUT*2 + 255) & ~(size_t)255; // f32 kvp[NBLK1][16384]
static constexpr size_t WS_KSP  = WS_KVP + (size_t)NBLK1*128*128*4;               // f32 ksp[NBLK1*4][128]

__device__ __forceinline__ unsigned f2bf1(float f) {            // f32 -> bf16 (RNE)
    unsigned u = __float_as_uint(f);
    return (u + 0x7FFFu + ((u >> 16) & 1u)) >> 16;
}
__device__ __forceinline__ unsigned packbf(float lo, float hi) {
    return (f2bf1(hi) << 16) | (f2bf1(lo) & 0xFFFFu);
}
__device__ __forceinline__ float elup1(float x) {               // elu(x)+1
    return x > 0.f ? x + 1.f : __expf(x);
}
__device__ __forceinline__ v8f wmma_bf16(const Frag& a, const Frag& b, v8f c) {
    return __builtin_amdgcn_wmma_f32_16x16x32_bf16(false, a.v, false, b.v,
                                                   (short)0, c, false, false);
}

// ---------------- kernel 1: W -> Wt bf16, n-major (B-fragment friendly) -----
__global__ void __launch_bounds__(256)
k_prep(const float* __restrict__ WQ, const float* __restrict__ WK,
       const float* __restrict__ WV, unsigned short* __restrict__ Wt) {
    int tid = blockIdx.x * 256 + threadIdx.x;        // 3*512*128 = 196608 exact
    int w = tid >> 16, r = tid & 65535;
    int k = r >> 7, n = r & 127;
    const float* src = (w == 0) ? WQ : (w == 1) ? WK : WV;
    Wt[(size_t)w*65536 + (size_t)n*512 + k] = (unsigned short)f2bf1(src[(size_t)k*128 + n]);
}

// ---------------- kernel 2: Q/K/V GEMMs + K^T V split-K, persistent ---------
__global__ void __launch_bounds__(128)
k_phase1(const float* __restrict__ h, const unsigned short* __restrict__ Wt,
         unsigned short* __restrict__ Qw, float* __restrict__ kvp,
         float* __restrict__ ksp) {
    __shared__ unsigned short sKt[128 * LDSTR];      // K^T tile: [i][m_local]
    __shared__ unsigned short sVt[128 * LDSTR];      // V^T tile: [j][m_local]

    const int wv = threadIdx.x >> 5, lane = threadIdx.x & 31;
    const int lh = lane & 15, sel = lane >> 4;

    v8f  kv[2][8] = {};                              // 128 VGPR KV accumulators
    float ks[8]   = {};                              // k_sum partials (col = nt*16+lh)

    for (int ch = blockIdx.x; ch < NCHUNK; ch += NBLK1) {
        const int m0 = ch * CHUNK + wv * 16;

        for (int w = 0; w < 3; ++w) {                // Q, K, V
            v8f c[8] = {};
            for (int kb = 0; kb < 16; ++kb) {        // K-dim 512 in steps of 32
                // A-frag from global h (fp32 -> bf16 inline), exact 16x32 layout
                const float* hp = h + (size_t)(m0 + lh) * D_IN + kb * 32 + sel * 8;
                f4 x0 = *(const f4*)hp,       x1 = *(const f4*)(hp + 4);
                f4 x2 = *(const f4*)(hp + 16), x3 = *(const f4*)(hp + 20);
                Frag a;
                a.u[0] = ui4{packbf(x0.x,x0.y), packbf(x0.z,x0.w),
                             packbf(x1.x,x1.y), packbf(x1.z,x1.w)};
                a.u[1] = ui4{packbf(x2.x,x2.y), packbf(x2.z,x2.w),
                             packbf(x3.x,x3.y), packbf(x3.z,x3.w)};
                #pragma unroll
                for (int nt = 0; nt < 8; ++nt) {     // 8 N-tiles of 16
                    const unsigned short* bp =
                        Wt + (size_t)(w*128 + nt*16 + lh)*512 + kb*32 + sel*8;
                    Frag b; b.u[0] = *(const ui4*)bp; b.u[1] = *(const ui4*)(bp + 16);
                    c[nt] = wmma_bf16(a, b, c[nt]);
                }
            }
            // epilogues (C layout: elem v -> row m0+v+sel*8, col nt*16+lh)
            if (w == 0) {                            // Q: elu+1, bf16, row-major
                #pragma unroll
                for (int nt = 0; nt < 8; ++nt)
                    #pragma unroll
                    for (int v = 0; v < 8; ++v)
                        Qw[(size_t)(m0 + v + sel*8)*128 + nt*16 + lh] =
                            (unsigned short)f2bf1(elup1(c[nt][v]));
            } else if (w == 1) {                     // K: elu+1, transpose to LDS
                #pragma unroll
                for (int nt = 0; nt < 8; ++nt)
                    #pragma unroll
                    for (int v = 0; v < 8; ++v) {
                        float kq = elup1(c[nt][v]);
                        ks[nt] += kq;
                        sKt[(nt*16 + lh)*LDSTR + wv*16 + v + sel*8] =
                            (unsigned short)f2bf1(kq);
                    }
            } else {                                 // V: transpose to LDS
                #pragma unroll
                for (int nt = 0; nt < 8; ++nt)
                    #pragma unroll
                    for (int v = 0; v < 8; ++v)
                        sVt[(nt*16 + lh)*LDSTR + wv*16 + v + sel*8] =
                            (unsigned short)f2bf1(c[nt][v]);
            }
        }
        __syncthreads();
        // KV += Kt(128 x 64) @ V(64 x 128): wave owns i-strips {2wv, 2wv+1}
        #pragma unroll
        for (int s = 0; s < 2; ++s) {
            const int i0 = (wv*2 + s) * 16;
            #pragma unroll
            for (int mb = 0; mb < 2; ++mb) {
                const int mo = mb * 32;
                const unsigned short* ap = &sKt[(i0 + lh)*LDSTR + mo + sel*8];
                Frag a; a.u[0] = *(const ui4*)ap; a.u[1] = *(const ui4*)(ap + 16);
                #pragma unroll
                for (int jt = 0; jt < 8; ++jt) {
                    const unsigned short* bp = &sVt[(jt*16 + lh)*LDSTR + mo + sel*8];
                    Frag b; b.u[0] = *(const ui4*)bp; b.u[1] = *(const ui4*)(bp + 16);
                    kv[s][jt] = wmma_bf16(a, b, kv[s][jt]);
                }
            }
        }
        __syncthreads();                             // LDS reused next chunk
    }
    // flush per-block partials (deterministic, no atomics)
    float* kvo = kvp + (size_t)blockIdx.x * 16384;
    #pragma unroll
    for (int s = 0; s < 2; ++s)
        #pragma unroll
        for (int jt = 0; jt < 8; ++jt)
            #pragma unroll
            for (int v = 0; v < 8; ++v)
                kvo[(size_t)((wv*2+s)*16 + v + sel*8)*128 + jt*16 + lh] = kv[s][jt][v];
    #pragma unroll
    for (int nt = 0; nt < 8; ++nt) {
        float t = ks[nt] + __shfl_xor(ks[nt], 16, 32);   // combine lane halves
        if (lane < 16)
            ksp[(size_t)(blockIdx.x*4 + wv)*128 + nt*16 + lh] = t;
    }
}

// ---------------- kernel 3: reduce partials -> KVt bf16 (n-major) + ksum ----
__global__ void __launch_bounds__(256)
k_reduce(const float* __restrict__ kvp, const float* __restrict__ ksp,
         unsigned short* __restrict__ KVt, float* __restrict__ ksum) {
    int tid = blockIdx.x * 256 + threadIdx.x;        // 16384 exact
    int i = tid >> 7, j = tid & 127;
    float acc = 0.f;
    for (int b = 0; b < NBLK1; ++b) acc += kvp[(size_t)b*16384 + tid];
    KVt[(size_t)j*128 + i] = (unsigned short)f2bf1(acc);   // transposed for B-frags
    if (tid < 128) {
        float a2 = 0.f;
        for (int b = 0; b < NBLK1*4; ++b) a2 += ksp[(size_t)b*128 + tid];
        ksum[tid] = a2;
    }
}

// ---------------- kernel 4: Z = 1 / (Q . ksum) ------------------------------
__global__ void __launch_bounds__(256)
k_z(const unsigned short* __restrict__ Qw, const float* __restrict__ ksum,
    float* __restrict__ Z) {
    int tid = blockIdx.x * 256 + threadIdx.x;
    if (tid >= N_ROWS) return;
    const ui4* qp = (const ui4*)(Qw + (size_t)tid * 128);
    float acc = 0.f;
    #pragma unroll
    for (int t = 0; t < 16; ++t) {
        ui4 q = qp[t];
        #pragma unroll
        for (int c2 = 0; c2 < 4; ++c2) {
            unsigned u = q[c2];
            float lo = __uint_as_float(u << 16);
            float hi = __uint_as_float(u & 0xFFFF0000u);
            int e = t*8 + c2*2;
            acc += lo * ksum[e] + hi * ksum[e + 1];  // ksum loads are wave-uniform
        }
    }
    Z[tid] = 1.f / acc;
}

// ---------------- kernel 5: out = elu((Q @ KV) * Z) -------------------------
__global__ void __launch_bounds__(128)
k_out(const unsigned short* __restrict__ Qw, const unsigned short* __restrict__ KVt,
      const float* __restrict__ Z, float* __restrict__ out) {
    const int wv = threadIdx.x >> 5, lane = threadIdx.x & 31;
    const int lh = lane & 15, sel = lane >> 4;
    const int m0 = blockIdx.x * CHUNK + wv * 16;
    v8f c[8] = {};
    #pragma unroll
    for (int kb = 0; kb < 4; ++kb) {                 // K-dim 128
        const unsigned short* ap = Qw + (size_t)(m0 + lh)*128 + kb*32 + sel*8;
        Frag a; a.u[0] = *(const ui4*)ap; a.u[1] = *(const ui4*)(ap + 16);
        #pragma unroll
        for (int jt = 0; jt < 8; ++jt) {
            const unsigned short* bp = KVt + (size_t)(jt*16 + lh)*128 + kb*32 + sel*8;
            Frag b; b.u[0] = *(const ui4*)bp; b.u[1] = *(const ui4*)(bp + 16);
            c[jt] = wmma_bf16(a, b, c[jt]);
        }
    }
    float zv[8];
    #pragma unroll
    for (int v = 0; v < 8; ++v) zv[v] = Z[m0 + v + sel*8];
    #pragma unroll
    for (int jt = 0; jt < 8; ++jt)
        #pragma unroll
        for (int v = 0; v < 8; ++v) {
            float y = c[jt][v] * zv[v];
            out[(size_t)(m0 + v + sel*8)*128 + jt*16 + lh] =
                y > 0.f ? y : __expf(y) - 1.f;
        }
}

// ---------------------------------------------------------------------------
extern "C" void kernel_launch(void* const* d_in, const int* in_sizes, int n_in,
                              void* d_out, int out_size, void* d_ws, size_t ws_size,
                              hipStream_t stream) {
    (void)in_sizes; (void)n_in; (void)out_size; (void)ws_size;
    const float* h  = (const float*)d_in[0];
    const float* WQ = (const float*)d_in[1];
    const float* WK = (const float*)d_in[2];
    const float* WV = (const float*)d_in[3];
    float* out = (float*)d_out;
    char*  ws  = (char*)d_ws;

    unsigned short* Wt   = (unsigned short*)(ws + WS_WT);
    unsigned short* KVt  = (unsigned short*)(ws + WS_KVT);
    float*          ksum = (float*)(ws + WS_KSUM);
    float*          Z    = (float*)(ws + WS_Z);
    unsigned short* Qw   = (unsigned short*)(ws + WS_Q);
    float*          kvp  = (float*)(ws + WS_KVP);
    float*          ksp  = (float*)(ws + WS_KSP);

    k_prep  <<<768, 256, 0, stream>>>(WQ, WK, WV, Wt);
    k_phase1<<<NBLK1, 128, 0, stream>>>(h, Wt, Qw, kvp, ksp);
    k_reduce<<<64, 256, 0, stream>>>(kvp, ksp, KVt, ksum);
    k_z     <<<(N_ROWS + 255) / 256, 256, 0, stream>>>(Qw, ksum, Z);
    k_out   <<<NCHUNK, 128, 0, stream>>>(Qw, KVt, Z, out);
}
